// SuperpointDiscriminativeLoss_72310069395901
// MI455X (gfx1250) — matrix operand order repaired and checked by reading, not compiled, not run on hardware
//
#include <hip/hip_runtime.h>

typedef __attribute__((ext_vector_type(16))) _Float16 v16h;
typedef __attribute__((ext_vector_type(8)))  float    v8f;

#define KCLS 13
#define NSP  50000
#define NGRP (NSP * KCLS)
#define CDIM 32
#define EPS_N 1e-12f
#define M_PULL 0.01f
#define M_PUSH 0.2f

__device__ __forceinline__ float waveSum32(float v) {
#pragma unroll
    for (int o = 16; o > 0; o >>= 1) v += __shfl_xor(v, o, 32);
    return v;
}

// ---------------- Pass 1: normalize points + scatter into group sums -------
__global__ __launch_bounds__(256) void spd_scatter(
    const float* __restrict__ p_fea, const int* __restrict__ sp_ids,
    const int* __restrict__ p_labels, float* __restrict__ sums,
    float* __restrict__ cnt, int* __restrict__ sp_count, int N)
{
    int lane = threadIdx.x & 31;
    int p = blockIdx.x * 8 + (threadIdx.x >> 5);
    if (p >= N) return;

    float v = p_fea[(size_t)p * CDIM + lane];
    float ss = waveSum32(v * v);
    float pn = v * (1.0f / fmaxf(sqrtf(ss), EPS_N));

    int lblr = p_labels[p];
    bool valid = (lblr != -1);
    int lbl = min(max(lblr, 0), KCLS - 1);
    int sp = sp_ids[p];
    size_t gid = (size_t)sp * KCLS + lbl;

    if (valid) {
        atomicAdd(&sums[gid * CDIM + lane], pn);
        if (lane == 0) atomicAdd(&cnt[gid], 1.0f);
    }
    if (lane == 0) atomicAdd(&sp_count[sp], 1);
}

// ---------------- Pass 2: centroid mean + normalize -> z (f32) and z_h (f16)
__global__ __launch_bounds__(256) void spd_centroid(
    float* __restrict__ sums, const float* __restrict__ cnt,
    _Float16* __restrict__ z_h)
{
    int lane = threadIdx.x & 31;
    int g = blockIdx.x * 8 + (threadIdx.x >> 5);
    if (g >= NGRP) return;

    float s = sums[(size_t)g * CDIM + lane];
    float c = cnt[g];
    float m = s / fmaxf(c, 1.0f);
    float ss = waveSum32(m * m);
    float z = m * (1.0f / fmaxf(sqrtf(ss), EPS_N));
    sums[(size_t)g * CDIM + lane] = z;          // in place: sums now holds z
    z_h[(size_t)g * CDIM + lane] = (_Float16)z;
}

// ---------------- Pass 3: exclusive prefix sum over sp_count ---------------
__global__ __launch_bounds__(1024) void spd_scan(
    const int* __restrict__ sp_count, int* __restrict__ sp_offset)
{
    __shared__ int sm[1024];
    __shared__ int carry;
    if (threadIdx.x == 0) carry = 0;
    __syncthreads();
    for (int base = 0; base < NSP; base += 1024) {
        int i = base + (int)threadIdx.x;
        int v = (i < NSP) ? sp_count[i] : 0;
        sm[threadIdx.x] = v;
        __syncthreads();
        for (int o = 1; o < 1024; o <<= 1) {
            int t = (threadIdx.x >= (unsigned)o) ? sm[threadIdx.x - o] : 0;
            __syncthreads();
            sm[threadIdx.x] += t;
            __syncthreads();
        }
        int incl = sm[threadIdx.x];
        int c = carry;
        if (i < NSP) sp_offset[i + 1] = c + incl;
        __syncthreads();
        if (threadIdx.x == 1023) carry = c + sm[1023];
        __syncthreads();
    }
    if (threadIdx.x == 0) sp_offset[0] = 0;
}

// ---------------- Pass 4: bin point indices by superpoint ------------------
__global__ __launch_bounds__(256) void spd_binpoints(
    const int* __restrict__ sp_ids, const int* __restrict__ sp_offset,
    int* __restrict__ sp_cursor, int* __restrict__ order, int N)
{
    int p = blockIdx.x * blockDim.x + threadIdx.x;
    if (p >= N) return;
    int sp = sp_ids[p];
    int pos = atomicAdd(&sp_cursor[sp], 1);
    order[sp_offset[sp] + pos] = p;
}

// ---------------- Pass 5: pull term (exact f32) ----------------------------
__global__ __launch_bounds__(256) void spd_pull(
    const float* __restrict__ p_fea, const int* __restrict__ sp_ids,
    const int* __restrict__ p_labels, const float* __restrict__ z,
    float* __restrict__ acc, int N)
{
    int lane = threadIdx.x & 31;
    int p = blockIdx.x * 8 + (threadIdx.x >> 5);
    if (p >= N) return;

    float v = p_fea[(size_t)p * CDIM + lane];
    float ss = waveSum32(v * v);
    float pn = v * (1.0f / fmaxf(sqrtf(ss), EPS_N));

    int lblr = p_labels[p];
    bool valid = (lblr != -1);
    int lbl = min(max(lblr, 0), KCLS - 1);
    size_t gid = (size_t)sp_ids[p] * KCLS + lbl;

    float d = pn - z[gid * CDIM + lane];
    float dd = waveSum32(d * d);
    if (lane == 0 && valid) {
        float l2 = sqrtf(dd + EPS_N);
        float t = fmaxf(l2 - M_PULL, 0.0f);
        atomicAdd(&acc[0], t * t);
        atomicAdd(&acc[1], 1.0f);
    }
}

// ---------------- Pass 6: push term via WMMA (wave per superpoint) ---------
__global__ __launch_bounds__(256) void spd_push_wmma(
    const float* __restrict__ p_fea, const int* __restrict__ p_labels,
    const _Float16* __restrict__ z_h, const float* __restrict__ cnt,
    const int* __restrict__ sp_offset, const int* __restrict__ order,
    float* __restrict__ acc)
{
    int lane = threadIdx.x & 31;
    int sp = blockIdx.x * 8 + (threadIdx.x >> 5);
    if (sp >= NSP) return;

    int hi = lane >> 4;          // K-chunk selector
    int lo = lane & 15;          // row M (A) / col N (B,D)

    int base = sp_offset[sp];
    int npts = sp_offset[sp + 1] - base;

    // B matrix: column N = lo holds class lo (zeros for lo >= 13).
    // Lane layout: lanes 0-15 hold K=0..15, lanes 16-31 hold K=16..31.
    v16h bmat;
#pragma unroll
    for (int e = 0; e < 16; ++e) bmat[e] = (_Float16)0.0f;
    bool colActive = (lo < KCLS);
    bool nonempty = false;
    if (colActive) {
        const _Float16* zp = z_h + ((size_t)sp * KCLS + lo) * CDIM + hi * 16;
        bmat = *(const v16h*)zp;                    // 32B aligned
        nonempty = cnt[(size_t)sp * KCLS + lo] > 0.0f;
    }

    float pushAcc = 0.0f, emAcc = 0.0f;

    for (int t = 0; t * 16 < npts; ++t) {
        int m = lo;
        bool rowValid = (t * 16 + m) < npts;
        int p = rowValid ? order[base + t * 16 + m] : 0;
        int lblr = rowValid ? p_labels[p] : -1;
        int pvalid = (rowValid && lblr != -1) ? 1 : 0;
        int plbl = min(max(lblr, 0), KCLS - 1);

        // A row M=lo: this lane covers K = 16*hi..16*hi+15
        float vals[16];
#pragma unroll
        for (int e = 0; e < 16; ++e) vals[e] = 0.0f;
        if (rowValid) {
            const float* row = p_fea + (size_t)p * CDIM;
            float4 a0 = *(const float4*)(row + 16 * hi);
            float4 a1 = *(const float4*)(row + 16 * hi + 4);
            float4 b0 = *(const float4*)(row + 16 * hi + 8);
            float4 b1 = *(const float4*)(row + 16 * hi + 12);
            vals[0]=a0.x; vals[1]=a0.y; vals[2]=a0.z; vals[3]=a0.w;
            vals[4]=a1.x; vals[5]=a1.y; vals[6]=a1.z; vals[7]=a1.w;
            vals[8]=b0.x; vals[9]=b0.y; vals[10]=b0.z; vals[11]=b0.w;
            vals[12]=b1.x; vals[13]=b1.y; vals[14]=b1.z; vals[15]=b1.w;
        }
        float sq = 0.0f;
#pragma unroll
        for (int e = 0; e < 16; ++e) sq += vals[e] * vals[e];
        float tot = sq + __shfl_xor(sq, 16, 32);    // pair lane covers other half
        float scale = 1.0f / fmaxf(sqrtf(tot), EPS_N);

        v16h amat;
#pragma unroll
        for (int e = 0; e < 16; ++e) amat[e] = (_Float16)(vals[e] * scale);

        v8f c = {};
        v8f d = __builtin_amdgcn_wmma_f32_16x16x32_f16(
            false, amat, false, bmat, (short)0, c, false, false);

        // D layout: d[r] at (M = r + 8*hi, N = lo)
#pragma unroll
        for (int r = 0; r < 8; ++r) {
            int M = r + 8 * hi;
            int lblM = __shfl(plbl, M, 32);
            int pvM  = __shfl(pvalid, M, 32);
            bool mask = colActive && nonempty && pvM && (lblM != lo);
            if (mask) {
                float e = fmaxf(d[r] - M_PUSH, 0.0f);
                pushAcc += e * e;
                emAcc += 1.0f;
            }
        }
    }

    pushAcc = waveSum32(pushAcc);
    emAcc = waveSum32(emAcc);
    if (lane == 0) {
        atomicAdd(&acc[2], pushAcc);
        atomicAdd(&acc[3], emAcc);
    }
}

// ---------------- Pass 7: finalize -----------------------------------------
__global__ void spd_finalize(const float* __restrict__ acc, float* __restrict__ out)
{
    out[0] = acc[0] / fmaxf(acc[1], 1.0f) + acc[2] / fmaxf(acc[3], 1.0f);
}

extern "C" void kernel_launch(void* const* d_in, const int* in_sizes, int n_in,
                              void* d_out, int out_size, void* d_ws, size_t ws_size,
                              hipStream_t stream)
{
    const float* p_fea = (const float*)d_in[0];
    const int* sp_ids = (const int*)d_in[1];
    const int* p_labels = (const int*)d_in[2];
    int N = in_sizes[0] / CDIM;

    // Workspace layout (zeroed regions first, single memset)
    char* w = (char*)d_ws;
    size_t off = 0;
    auto take = [&](size_t bytes) -> char* {
        char* p = w + off;
        off = (off + bytes + 255) & ~(size_t)255;
        return p;
    };
    float* sums     = (float*)take((size_t)NGRP * CDIM * 4);  // becomes z (f32)
    float* cnt      = (float*)take((size_t)NGRP * 4);
    int*   sp_count = (int*)take((size_t)NSP * 4);
    int*   sp_cursor= (int*)take((size_t)NSP * 4);
    float* acc      = (float*)take(4 * 4);
    size_t zero_bytes = off;
    int*   sp_offset = (int*)take((size_t)(NSP + 1) * 4);
    _Float16* z_h   = (_Float16*)take((size_t)NGRP * CDIM * 2);
    int*   order    = (int*)take((size_t)N * 4);
    (void)ws_size; (void)n_in; (void)out_size;

    hipMemsetAsync(d_ws, 0, zero_bytes, stream);

    int blkPts = (N + 7) / 8;           // wave per point, 8 waves/block
    spd_scatter<<<blkPts, 256, 0, stream>>>(p_fea, sp_ids, p_labels,
                                            sums, cnt, sp_count, N);

    int blkGrp = (NGRP + 7) / 8;        // wave per group
    spd_centroid<<<blkGrp, 256, 0, stream>>>(sums, cnt, z_h);

    spd_scan<<<1, 1024, 0, stream>>>(sp_count, sp_offset);

    spd_binpoints<<<(N + 255) / 256, 256, 0, stream>>>(sp_ids, sp_offset,
                                                       sp_cursor, order, N);

    spd_pull<<<blkPts, 256, 0, stream>>>(p_fea, sp_ids, p_labels, sums, acc, N);

    int blkSp = (NSP + 7) / 8;          // wave per superpoint
    spd_push_wmma<<<blkSp, 256, 0, stream>>>(p_fea, p_labels, z_h, cnt,
                                             sp_offset, order, acc);

    spd_finalize<<<1, 1, 0, stream>>>(acc, (float*)d_out);
}
